// GATLayer_25993142075918
// MI455X (gfx1250) — matrix operand order repaired and checked by reading, not compiled, not run on hardware
//
#include <hip/hip_runtime.h>
#include <hip/hip_bf16.h>

// ---------------------------------------------------------------------------
// GAT layer, fused flash-softmax formulation for MI455X (gfx1250, wave32).
//   h  = x @ W                      (8192x64)
//   s1 = h @ a[:64], s2 = h @ a[64:]
//   e  = leakyrelu(s1[i]+s2[j]); masked by adj; softmax rows; out = elu(att@h)
// adj (256 MB) is streamed exactly once; att (8192^2) is never materialized.
// The PV matmul uses v_wmma_f32_16x16x32_f16 with f32 accumulation.
// ---------------------------------------------------------------------------

typedef __attribute__((ext_vector_type(16))) _Float16 v16h;
typedef __attribute__((ext_vector_type(8)))  float    v8f;

#define GAT_N     8192
#define GAT_FIN   128
#define GAT_FOUT  64
#define GAT_ALPHA 0.2f
#define GAT_MASK  (-9000000000000000.0f)

// ---------------------------------------------------------------------------
// Kernel 1: h = x@W (f32), store h transposed as f16 (B fragment loads
// read 16 contiguous K per lane), plus s1 = h.a1, s2 = h.a2.  134 MFLOP -> trivial.
// One block (64 threads) per row.
// ---------------------------------------------------------------------------
__global__ void gat_h_kernel(const float* __restrict__ x,
                             const float* __restrict__ W,
                             const float* __restrict__ a,
                             _Float16* __restrict__ hT,   // [64][8192] f16
                             float* __restrict__ s1,
                             float* __restrict__ s2) {
  __shared__ float xs[GAT_FIN];
  __shared__ float r1[GAT_FOUT];
  __shared__ float r2[GAT_FOUT];
  const int row = blockIdx.x;
  const int t   = threadIdx.x;

  xs[t]      = x[(size_t)row * GAT_FIN + t];
  xs[t + 64] = x[(size_t)row * GAT_FIN + 64 + t];
  __syncthreads();

  float acc = 0.f;
#pragma unroll 8
  for (int k = 0; k < GAT_FIN; ++k)
    acc = fmaf(xs[k], W[k * GAT_FOUT + t], acc);

  hT[(size_t)t * GAT_N + row] = (_Float16)acc;
  r1[t] = acc * a[t];
  r2[t] = acc * a[GAT_FOUT + t];
  __syncthreads();
  if (t == 0) {
    float v1 = 0.f, v2 = 0.f;
    for (int i = 0; i < GAT_FOUT; ++i) { v1 += r1[i]; v2 += r2[i]; }
    s1[row] = v1;
    s2[row] = v2;
  }
}

// ---------------------------------------------------------------------------
// Kernel 2: one-pass flash softmax + WMMA PV matmul.
// One wave32 per 16-row tile per column split. 4 waves per block.
// Per K-step of 32 cols:
//   - lane computes its 16 e values already in contiguous-K A-fragment order
//     (lane 0-15: M=lane, K=0-15; lane 16-31: K=16-31)
//   - online softmax (m,l replicated in lane r and r+16; one shfl_xor(16))
//   - acc *= exp(m_old-m_new) (row broadcast via 8 shfls; C-layout M=j+8*hi)
//   - 4x v_wmma_f32_16x16x32_f16: acc[16x16 per N-tile] += P(16x32) x hT(32x16)
// ---------------------------------------------------------------------------
__global__ void gat_flash_kernel(const int* __restrict__ adj,
                                 const _Float16* __restrict__ hT,
                                 const float* __restrict__ s1,
                                 const float* __restrict__ s2,
                                 float* __restrict__ Pm,
                                 float* __restrict__ Pl,
                                 float* __restrict__ Pacc,
                                 int nsplit) {
  const int lane  = threadIdx.x & 31;
  const int wave  = threadIdx.x >> 5;
  const int tile  = blockIdx.x * 4 + wave;        // 0..511
  const int split = blockIdx.y;
  const int cols_per = GAT_N / nsplit;
  const int cbeg = split * cols_per;
  const int cend = cbeg + cols_per;

  const int hi   = lane >> 4;                     // K-half select
  const int row  = lane & 15;                     // local M row
  const int grow = tile * 16 + row;
  const int ncol = lane & 15;                     // N column inside 16-tile

  const float s1r = s1[grow];
  const int* adjRow = adj + (size_t)grow * GAT_N;

  v8f acc0 = {}, acc1 = {}, acc2 = {}, acc3 = {};
  float m = -__builtin_inff();
  float l = 0.f;

  for (int c0 = cbeg; c0 < cend; c0 += 32) {
    const int b0 = c0 + hi * 16;                  // first 8-col run (K = 16*hi + 0..7)
    const int b1 = c0 + hi * 16 + 8;              // second 8-col run (K = 16*hi + 8..15)

    if (c0 + 160 < cend) {                        // stream-ahead on adj
      __builtin_prefetch(adjRow + b0 + 128, 0, 1);
      __builtin_prefetch(adjRow + b1 + 128, 0, 1);
    }

    const int4   A0 = *(const int4*)(adjRow + b0);
    const int4   A1 = *(const int4*)(adjRow + b0 + 4);
    const int4   A2 = *(const int4*)(adjRow + b1);
    const int4   A3 = *(const int4*)(adjRow + b1 + 4);
    const float4 S0 = *(const float4*)(s2 + b0);
    const float4 S1 = *(const float4*)(s2 + b0 + 4);
    const float4 S2 = *(const float4*)(s2 + b1);
    const float4 S3 = *(const float4*)(s2 + b1 + 4);

    const int   av[16] = {A0.x, A0.y, A0.z, A0.w, A1.x, A1.y, A1.z, A1.w,
                          A2.x, A2.y, A2.z, A2.w, A3.x, A3.y, A3.z, A3.w};
    const float sv[16] = {S0.x, S0.y, S0.z, S0.w, S1.x, S1.y, S1.z, S1.w,
                          S2.x, S2.y, S2.z, S2.w, S3.x, S3.y, S3.z, S3.w};

    float p[16];
    float tmax = -__builtin_inff();
#pragma unroll
    for (int i = 0; i < 16; ++i) {
      const float pre = s1r + sv[i];
      float e = pre > 0.f ? pre : GAT_ALPHA * pre;  // leaky_relu
      e = (av[i] > 0) ? e : GAT_MASK;               // adjacency mask
      p[i] = e;
      tmax = fmaxf(tmax, e);
    }
    tmax = fmaxf(tmax, __shfl_xor(tmax, 16, 32));   // join row halves

    const float mnew  = fmaxf(m, tmax);
    const float scale = __expf(m - mnew);           // first iter: exp(-inf)=0

    float tsum = 0.f;
#pragma unroll
    for (int i = 0; i < 16; ++i) {
      const float q = __expf(p[i] - mnew);
      p[i] = q;
      tsum += q;
    }
    tsum += __shfl_xor(tsum, 16, 32);
    l = l * scale + tsum;
    m = mnew;

    // rescale accumulators; C-layout: VGPR j holds M = j + 8*hi
#pragma unroll
    for (int j = 0; j < 8; ++j) {
      const float rs = __shfl(scale, (hi << 3) + j, 32);
      acc0[j] *= rs; acc1[j] *= rs; acc2[j] *= rs; acc3[j] *= rs;
    }

    // A fragment: p[] is already in contiguous-K order, same as B
    v16h af;
#pragma unroll
    for (int i = 0; i < 16; ++i) af[i] = (_Float16)p[i];

    // B fragments: hT rows n, K = c0 + hi*16 .. +15 contiguous (32B aligned)
    const _Float16* hb = hT + c0 + hi * 16;
    const v16h bf0 = *(const v16h*)(hb + (size_t)(ncol)      * GAT_N);
    const v16h bf1 = *(const v16h*)(hb + (size_t)(ncol + 16) * GAT_N);
    const v16h bf2 = *(const v16h*)(hb + (size_t)(ncol + 32) * GAT_N);
    const v16h bf3 = *(const v16h*)(hb + (size_t)(ncol + 48) * GAT_N);

    acc0 = __builtin_amdgcn_wmma_f32_16x16x32_f16(false, af, false, bf0,
                                                  (short)0, acc0, false, false);
    acc1 = __builtin_amdgcn_wmma_f32_16x16x32_f16(false, af, false, bf1,
                                                  (short)0, acc1, false, false);
    acc2 = __builtin_amdgcn_wmma_f32_16x16x32_f16(false, af, false, bf2,
                                                  (short)0, acc2, false, false);
    acc3 = __builtin_amdgcn_wmma_f32_16x16x32_f16(false, af, false, bf3,
                                                  (short)0, acc3, false, false);
  }

  // ---- store partial results for this column split ----
  if (hi == 0) {
    Pm[(size_t)split * GAT_N + grow] = m;
    Pl[(size_t)split * GAT_N + grow] = l;
  }
  float* accbase = Pacc + ((size_t)split * GAT_N + (size_t)tile * 16) * GAT_FOUT;
#pragma unroll
  for (int j = 0; j < 8; ++j) {
    const int r = j + 8 * hi;                      // C-layout row
    float* q = accbase + (size_t)r * GAT_FOUT + ncol;
    q[0]  = acc0[j];
    q[16] = acc1[j];
    q[32] = acc2[j];
    q[48] = acc3[j];
  }
}

// ---------------------------------------------------------------------------
// Kernel 3: exact merge of the column-split partials + ELU epilogue.
// out[r][c] = elu( sum_s e^{m_s-M} acc_s[r][c] / sum_s e^{m_s-M} l_s[r] )
// ---------------------------------------------------------------------------
__global__ void gat_merge_kernel(const float* __restrict__ Pm,
                                 const float* __restrict__ Pl,
                                 const float* __restrict__ Pacc,
                                 float* __restrict__ out,
                                 int nsplit) {
  const int row = blockIdx.x;
  const int col = threadIdx.x;
  float M = -__builtin_inff();
  for (int s = 0; s < nsplit; ++s)
    M = fmaxf(M, Pm[(size_t)s * GAT_N + row]);
  float num = 0.f, den = 0.f;
  for (int s = 0; s < nsplit; ++s) {
    const float w = __expf(Pm[(size_t)s * GAT_N + row] - M);
    den += w * Pl[(size_t)s * GAT_N + row];
    num += w * Pacc[((size_t)s * GAT_N + row) * GAT_FOUT + col];
  }
  const float v = num / den;
  out[(size_t)row * GAT_FOUT + col] = v > 0.f ? v : (__expf(v) - 1.f); // ELU
}

// ---------------------------------------------------------------------------
extern "C" void kernel_launch(void* const* d_in, const int* in_sizes, int n_in,
                              void* d_out, int out_size, void* d_ws, size_t ws_size,
                              hipStream_t stream) {
  const float* x   = (const float*)d_in[0];
  const int*   adj = (const int*)d_in[1];
  const float* W   = (const float*)d_in[2];
  const float* a   = (const float*)d_in[3];
  float*       out = (float*)d_out;

  char* ws = (char*)d_ws;
  size_t off = 0;
  _Float16* hT = (_Float16*)(ws + off); off += (size_t)GAT_FOUT * GAT_N * sizeof(_Float16);
  float*    s1 = (float*)(ws + off);    off += (size_t)GAT_N * sizeof(float);
  float*    s2 = (float*)(ws + off);    off += (size_t)GAT_N * sizeof(float);

  // choose column-split count from available workspace (deterministic)
  const size_t fixed = off;
  const size_t per_split = (size_t)GAT_N * sizeof(float) * 2      // m, l
                         + (size_t)GAT_N * GAT_FOUT * sizeof(float); // acc
  int nsplit = 8;
  while (nsplit > 1 && fixed + (size_t)nsplit * per_split > ws_size) nsplit >>= 1;

  float* Pm   = (float*)(ws + off); off += (size_t)nsplit * GAT_N * sizeof(float);
  float* Pl   = (float*)(ws + off); off += (size_t)nsplit * GAT_N * sizeof(float);
  float* Pacc = (float*)(ws + off);

  gat_h_kernel<<<GAT_N, 64, 0, stream>>>(x, W, a, hT, s1, s2);

  dim3 grid(GAT_N / 16 / 4, nsplit);              // 4 waves (tiles) per block
  gat_flash_kernel<<<grid, 128, 0, stream>>>(adj, hT, s1, s2, Pm, Pl, Pacc, nsplit);

  gat_merge_kernel<<<GAT_N, GAT_FOUT, 0, stream>>>(Pm, Pl, Pacc, out, nsplit);
}
